// Attention_60885456388891
// MI455X (gfx1250) — compile-verified, matrix operand order checked by reading, not run on hardware
//
#include <hip/hip_runtime.h>
#include <hip/hip_bf16.h>
#include <math.h>

typedef __attribute__((ext_vector_type(16))) _Float16 v16h;
typedef __attribute__((ext_vector_type(8)))  _Float16 v8h;
typedef __attribute__((ext_vector_type(4)))  _Float16 v4h;
typedef __attribute__((ext_vector_type(8)))  float    v8f;

#define BB 512
#define NN_NODES 128
#define NN_RELAS 256
#define DD 512
#define DD2 1024
#define DD4 128          // DD / 4 (float4 chunks per row)

// ---------------------------------------------------------------------------
// f32 -> f16 convert
// ---------------------------------------------------------------------------
__global__ void cvt_f16_kernel(const float* __restrict__ src,
                               _Float16* __restrict__ dst, int n) {
    int i = blockIdx.x * blockDim.x + threadIdx.x;
    if (i < n) dst[i] = (_Float16)src[i];
}

// Wt[j*K + k] = W[k*N + j]  (transpose + convert); W is [K, N]
__global__ void cvt_f16_transpose_kernel(const float* __restrict__ W,
                                         _Float16* __restrict__ Wt,
                                         int K, int N) {
    int i = blockIdx.x * blockDim.x + threadIdx.x;
    if (i >= K * N) return;
    int k = i / N;
    int j = i - k * N;
    Wt[(size_t)j * K + k] = (_Float16)W[i];
}

// ---------------------------------------------------------------------------
// WMMA fragment load: 16x32 f16 tile, ISA 7.12.2 layout.
// lanes 0-15: row M=lane,     K = {k0+0..7,  k0+16..23}
// lanes16-31: row M=lane-16,  K = {k0+8..15, k0+24..31}
// Identical layout used for B via the transposed weight matrix (Wt[N,K]).
// ---------------------------------------------------------------------------
__device__ __forceinline__ v16h load_frag16(const _Float16* __restrict__ base,
                                            int ld, int row0, int k0, int lane) {
    int r    = row0 + (lane & 15);
    int koff = (lane & 16) ? 8 : 0;
    const _Float16* p = base + (size_t)r * ld + k0 + koff;
    union { v16h v; v8h h[2]; } u;
    u.h[0] = *(const v8h*)(p);        // K: k0+koff   .. k0+koff+7
    u.h[1] = *(const v8h*)(p + 16);   // K: k0+16+koff.. k0+23+koff
    return u.v;
}

// ---------------------------------------------------------------------------
// WMMA GEMM: C[M,N] = A[M,K] (f16) @ Wt[N,K]^T (f16) + bias[N], f32 accumulate.
// One 16x16 tile per wave; blockDim=256 (8 waves); grid = tiles/8 (exact).
// ---------------------------------------------------------------------------
__global__ void wmma_gemm_bias_kernel(const _Float16* __restrict__ A,
                                      const _Float16* __restrict__ Bt,
                                      const float* __restrict__ bias,
                                      float* __restrict__ C,
                                      int M, int N, int K) {
    int wave = (blockIdx.x * blockDim.x + threadIdx.x) >> 5;
    int lane = threadIdx.x & 31;
    int tilesN = N >> 4;
    int tM = (wave / tilesN) << 4;
    int tN = (wave % tilesN) << 4;
    if (tM >= M) return;                 // uniform per wave: EXEC stays all-1s

    v8f c = {};
    for (int k = 0; k < K; k += 32) {
        v16h a = load_frag16(A,  K, tM, k, lane);
        v16h b = load_frag16(Bt, K, tN, k, lane);
        c = __builtin_amdgcn_wmma_f32_16x16x32_f16(
                false, a, false, b, (short)0, c, false, false);
    }

    // C/D layout: lane n = tN + (lane&15); VGPR v holds M = v (+8 if lane>=16)
    int n  = tN + (lane & 15);
    int m0 = tM + ((lane & 16) ? 8 : 0);
    float bv = bias[n];
#pragma unroll
    for (int v = 0; v < 8; ++v)
        C[(size_t)(m0 + v) * N + n] = c[v] + bv;
}

// ---------------------------------------------------------------------------
// Fused Bahdanau attention for one batch element per block (256 thr, 8 waves):
//   score_n = sum_d tanh(p[b,n,d] + hproj[b,d]) * walpha[d] + balpha
//   w = softmax(score) * mask / sum          (== masked softmax)
//   res[b,d] = sum_n w[n] * feats[b,n,d]     -> written as f16 into concat buf
// All global traffic uses b128 (float4) loads to saturate HBM.
// ---------------------------------------------------------------------------
__global__ void attn_kernel(const float* __restrict__ pfeats,  // [B,NN,512]
                            const float* __restrict__ feats,   // [B,NN,512]
                            const int*   __restrict__ masks,   // [B,NN]
                            const float* __restrict__ hproj,   // [B,512]
                            const float* __restrict__ walpha,  // [512]
                            const float* __restrict__ balpha,  // [1]
                            _Float16* __restrict__ out0,       // row stride 1024
                            _Float16* __restrict__ out1,       // optional
                            int NN) {
    __shared__ float4 sh_h4[DD4];
    __shared__ float4 sh_w4[DD4];
    __shared__ float  sc[NN_RELAS];      // max(N,R)
    __shared__ float4 red4[DD4];

    int b = blockIdx.x;
    int t = threadIdx.x;
    if (t < DD4)
        sh_h4[t] = ((const float4*)hproj)[(size_t)b * DD4 + t];
    else
        sh_w4[t - DD4] = ((const float4*)walpha)[t - DD4];
    __syncthreads();

    int wave = t >> 5, lane = t & 31;
    float ba = balpha[0];

    // ---- scores: each wave owns rows n = wave, wave+8, ... (b128 loads) ----
    for (int n = wave; n < NN; n += 8) {
        const float4* p4 = (const float4*)(pfeats + ((size_t)b * NN + n) * DD);
        float acc = 0.f;
#pragma unroll
        for (int i = 0; i < 4; ++i) {
            int d4 = lane + 32 * i;                   // coalesced 512B/wave
            float4 pv = p4[d4];
            float4 hv = sh_h4[d4];
            float4 wv = sh_w4[d4];
            acc += tanhf(pv.x + hv.x) * wv.x;
            acc += tanhf(pv.y + hv.y) * wv.y;
            acc += tanhf(pv.z + hv.z) * wv.z;
            acc += tanhf(pv.w + hv.w) * wv.w;
        }
#pragma unroll
        for (int off = 16; off; off >>= 1)
            acc += __shfl_xor(acc, off, 32);
        if (lane == 0) sc[n] = acc + ba;
    }
    __syncthreads();

    // ---- masked softmax (wave 0) ----
    if (wave == 0) {
        float m = -INFINITY;
        for (int n = lane; n < NN; n += 32) m = fmaxf(m, sc[n]);
#pragma unroll
        for (int off = 16; off; off >>= 1)
            m = fmaxf(m, __shfl_xor(m, off, 32));
        float s = 0.f;
        for (int n = lane; n < NN; n += 32) {
            float e = __expf(sc[n] - m) * (float)masks[(size_t)b * NN + n];
            sc[n] = e;
            s += e;
        }
#pragma unroll
        for (int off = 16; off; off >>= 1)
            s += __shfl_xor(s, off, 32);
        float inv = 1.f / s;
        for (int n = lane; n < NN; n += 32) sc[n] *= inv;
    }
    __syncthreads();

    // ---- weighted sum: thread owns float4 chunk d4 = t&127, n-parity t>>7 ----
    int d4   = t & (DD4 - 1);
    int half = t >> 7;                   // 0 or 1
    float4 acc = {0.f, 0.f, 0.f, 0.f};
#pragma unroll 2
    for (int n = half; n < NN; n += 2) {
        float w = sc[n];
        const float4* f4 = (const float4*)(feats + ((size_t)b * NN + n) * DD);
        float4 v = f4[d4];               // b128, row fully coalesced
        acc.x += w * v.x; acc.y += w * v.y;
        acc.z += w * v.z; acc.w += w * v.w;
    }
    if (half) red4[d4] = acc;
    __syncthreads();
    if (!half) {
        float4 o = red4[d4];
        acc.x += o.x; acc.y += o.y; acc.z += o.z; acc.w += o.w;
        v4h hv;
        hv[0] = (_Float16)acc.x; hv[1] = (_Float16)acc.y;
        hv[2] = (_Float16)acc.z; hv[3] = (_Float16)acc.w;
        *(v4h*)(out0 + (size_t)b * DD2 + 4 * d4) = hv;   // 8B packed store
        if (out1)
            *(v4h*)(out1 + (size_t)b * DD2 + 4 * d4) = hv;
    }
}

// ---------------------------------------------------------------------------
// GLU: out[b,j] = g[b,j] * sigmoid(g[b,j+512]); optional f16 copy (stride 1024)
// ---------------------------------------------------------------------------
__global__ void glu_kernel(const float* __restrict__ g,   // [B, 1024]
                           float* __restrict__ out_f32,   // [B, 512] flat
                           _Float16* __restrict__ out_f16) {
    int i = blockIdx.x * blockDim.x + threadIdx.x;        // over B*512
    if (i >= BB * DD) return;
    int col = i & (DD - 1);
    int row = i >> 9;
    float av = g[(size_t)row * DD2 + col];
    float bv = g[(size_t)row * DD2 + DD + col];
    float r  = av * (1.f / (1.f + __expf(-bv)));
    out_f32[i] = r;
    if (out_f16) out_f16[(size_t)row * DD2 + col] = (_Float16)r;
}

// ---------------------------------------------------------------------------
extern "C" void kernel_launch(void* const* d_in, const int* in_sizes, int n_in,
                              void* d_out, int out_size, void* d_ws, size_t ws_size,
                              hipStream_t stream) {
    const float* h        = (const float*)d_in[0];
    const float* nfeat    = (const float*)d_in[1];
    const float* pnfeat   = (const float*)d_in[2];
    const float* rfeat    = (const float*)d_in[3];
    const float* prfeat   = (const float*)d_in[4];
    const int*   amask    = (const int*)d_in[5];
    const int*   rmask    = (const int*)d_in[6];
    const float* W_h2n    = (const float*)d_in[7];
    const float* b_h2n    = (const float*)d_in[8];
    const float* W_h2r    = (const float*)d_in[9];
    const float* b_h2r    = (const float*)d_in[10];
    const float* w_a1     = (const float*)d_in[11];
    const float* b_a1     = (const float*)d_in[12];
    const float* w_a2     = (const float*)d_in[13];
    const float* b_a2     = (const float*)d_in[14];
    const float* W_ng     = (const float*)d_in[15];
    const float* b_ng     = (const float*)d_in[16];
    const float* W_rg     = (const float*)d_in[17];
    const float* b_rg     = (const float*)d_in[18];
    float* out = (float*)d_out;

    char* ws = (char*)d_ws;
    size_t off = 0;
    auto alloc = [&](size_t bytes) { char* p = ws + off; off += (bytes + 255) & ~(size_t)255; return p; };
    _Float16* h16    = (_Float16*)alloc((size_t)BB * DD * 2);
    _Float16* WtA    = (_Float16*)alloc((size_t)DD * DD * 2);      // h2node^T
    _Float16* WtB    = (_Float16*)alloc((size_t)DD * DD * 2);      // h2rela^T
    _Float16* WtNG   = (_Float16*)alloc((size_t)DD2 * DD2 * 2);
    _Float16* WtRG   = (_Float16*)alloc((size_t)DD2 * DD2 * 2);
    float*    node_h = (float*)   alloc((size_t)BB * DD * 4);
    float*    rela_h = (float*)   alloc((size_t)BB * DD * 4);
    _Float16* cat1   = (_Float16*)alloc((size_t)BB * DD2 * 2);     // [node16|rela16]
    _Float16* cat2   = (_Float16*)alloc((size_t)BB * DD2 * 2);     // [rela16|nres16]
    float*    g1     = (float*)   alloc((size_t)BB * DD2 * 4);
    float*    g2     = (float*)   alloc((size_t)BB * DD2 * 4);
    (void)ws_size; (void)n_in; (void)in_sizes; (void)out_size;

    // 1. conversions / transposes (tiny)
    cvt_f16_kernel<<<(BB * DD + 255) / 256, 256, 0, stream>>>(h, h16, BB * DD);
    cvt_f16_transpose_kernel<<<(DD * DD + 255) / 256, 256, 0, stream>>>(W_h2n, WtA, DD, DD);
    cvt_f16_transpose_kernel<<<(DD * DD + 255) / 256, 256, 0, stream>>>(W_h2r, WtB, DD, DD);
    cvt_f16_transpose_kernel<<<(DD2 * DD2 + 255) / 256, 256, 0, stream>>>(W_ng, WtNG, DD2, DD2);
    cvt_f16_transpose_kernel<<<(DD2 * DD2 + 255) / 256, 256, 0, stream>>>(W_rg, WtRG, DD2, DD2);

    // 2. projection GEMMs: [512,512] x [512,512]  (1024 tiles -> 128 blocks)
    wmma_gemm_bias_kernel<<<(BB / 16) * (DD / 16) / 8, 256, 0, stream>>>(
        h16, WtA, b_h2n, node_h, BB, DD, DD);
    wmma_gemm_bias_kernel<<<(BB / 16) * (DD / 16) / 8, 256, 0, stream>>>(
        h16, WtB, b_h2r, rela_h, BB, DD, DD);

    // 3. streaming attention (the HBM-bound part: ~810 MB total, all b128)
    attn_kernel<<<BB, 256, 0, stream>>>(pnfeat, nfeat, amask, node_h, w_a1, b_a1,
                                        cat1 /*cols 0..511*/, nullptr, NN_NODES);
    attn_kernel<<<BB, 256, 0, stream>>>(prfeat, rfeat, rmask, rela_h, w_a2, b_a2,
                                        cat1 + DD /*cols 512..1023*/,
                                        cat2 /*cols 0..511*/, NN_RELAS);

    // 4. fusion GEMM 1 + GLU -> node_res (also f16 into cat2 cols 512..1023)
    wmma_gemm_bias_kernel<<<(BB / 16) * (DD2 / 16) / 8, 256, 0, stream>>>(
        cat1, WtNG, b_ng, g1, BB, DD2, DD2);
    glu_kernel<<<(BB * DD + 255) / 256, 256, 0, stream>>>(g1, out, cat2 + DD);

    // 5. fusion GEMM 2 + GLU -> rela_res
    wmma_gemm_bias_kernel<<<(BB / 16) * (DD2 / 16) / 8, 256, 0, stream>>>(
        cat2, WtRG, b_rg, g2, BB, DD2, DD2);
    glu_kernel<<<(BB * DD + 255) / 256, 256, 0, stream>>>(g2, out + (size_t)BB * DD, nullptr);
}
